// ModifiedBert_57672820851396
// MI455X (gfx1250) — matrix operand order repaired and checked by reading, not compiled, not run on hardware
//
#include <hip/hip_runtime.h>
#include <hip/hip_bf16.h>
#include <cstdint>

typedef uint32_t u32;
typedef uint64_t u64;
typedef __hip_bfloat16 hbf;

typedef unsigned int uivec4 __attribute__((ext_vector_type(4)));
typedef int          ivec4  __attribute__((ext_vector_type(4)));
typedef int          ivec8  __attribute__((ext_vector_type(8)));
typedef float        fvec8  __attribute__((ext_vector_type(8)));
typedef __bf16       bvec16 __attribute__((ext_vector_type(16)));

union Frag { uivec4 u[2]; bvec16 v; };

#define WMMA_BF16(a, b, c) \
  __builtin_amdgcn_wmma_f32_16x16x32_bf16(false, (a), false, (b), (short)0, (c), false, false)

// ---------------------------------------------------------------------------
// Tensor Data Mover (TDM) 2D tile load: global (row-major, bf16) -> LDS with
// 16B padding after every 128B row (LDS pitch 144B, conflict-free frag reads).
// ---------------------------------------------------------------------------
#if defined(__gfx1250__) && __has_builtin(__builtin_amdgcn_tensor_load_to_lds) && \
    __has_builtin(__builtin_amdgcn_s_wait_tensorcnt)
#define HAVE_TDM 1
#else
#define HAVE_TDM 0
#endif

#if HAVE_TDM
__device__ __forceinline__ void tdm_load_2d(u32 lds_addr, const void* gptr,
                                            u32 tdim0, u32 tdim1,
                                            u32 tiled0, u32 tiled1, u64 stride0) {
  u64 ga = (u64)(uintptr_t)gptr;
  uivec4 g0;
  g0[0] = 1u;                                            // count=1, user mode
  g0[1] = lds_addr;                                      // LDS byte address
  g0[2] = (u32)ga;                                       // global_addr[31:0]
  g0[3] = (u32)((ga >> 32) & 0x1ffffffu) | (2u << 30);   // addr[56:32] | type=2
  ivec8 g1;
  // data_size=2B(code1); pad_enable; pad every 32 DWORDs (code 4) by 4 DWORDs (code 3)
  g1[0] = (int)((1u << 16) | (1u << 20) | (4u << 22) | (3u << 25));
  g1[1] = (int)((tdim0 & 0xffffu) << 16);                // tensor_dim0 lo16
  g1[2] = (int)((tdim0 >> 16) | ((tdim1 & 0xffffu) << 16));
  g1[3] = (int)((tdim1 >> 16) | (tiled0 << 16));         // tile_dim0
  g1[4] = (int)(tiled1 & 0xffffu);                       // tile_dim1 (tile_dim2=0)
  g1[5] = (int)(u32)stride0;                             // tensor_dim0_stride lo32
  g1[6] = (int)(u32)((stride0 >> 32) & 0xffffu);
  g1[7] = 0;
  ivec4 z4 = {};
#if __clang_major__ >= 23
  ivec8 z8 = {};
  __builtin_amdgcn_tensor_load_to_lds(g0, g1, z4, z4, z8, 0);
#else
  __builtin_amdgcn_tensor_load_to_lds(g0, g1, z4, z4, 0);
#endif
}
#endif

// ---------------------------------------------------------------------------
// GEMM: C[M,N] = A[M,K](bf16) * Bt[N,K](bf16)^T + bias  (M=4096 fixed)
//   OP 0: store bf16; OP 1: GELU(exact erf) -> bf16; OP 2: +res -> f32
// WG 256 thr, tile 128x64, K-step 64, double-buffered TDM LDS staging.
// ---------------------------------------------------------------------------
#define LDSP 144
#define ATILE (128 * LDSP)
#define BTILE (64 * LDSP)

__device__ __forceinline__ void gemm_tile_compute(const char* aB, const char* bB,
                                                  int wm, int wn, int li, int hi,
                                                  fvec8 (&acc)[2][2]) {
#pragma unroll
  for (int kc = 0; kc < 64; kc += 32) {
    Frag af[2];
#pragma unroll
    for (int tm = 0; tm < 2; ++tm) {
      const char* p = aB + (size_t)(wm * 32 + tm * 16 + li) * LDSP + (kc + hi * 16) * 2;
      af[tm].u[0] = *(const uivec4*)p;
      af[tm].u[1] = *(const uivec4*)(p + 16);
    }
#pragma unroll
    for (int tn = 0; tn < 2; ++tn) {
      Frag bfr;
      const char* p = bB + (size_t)(wn * 32 + tn * 16 + li) * LDSP + (kc + hi * 16) * 2;
      bfr.u[0] = *(const uivec4*)p;
      bfr.u[1] = *(const uivec4*)(p + 16);
#pragma unroll
      for (int tm = 0; tm < 2; ++tm)
        acc[tm][tn] = WMMA_BF16(af[tm].v, bfr.v, acc[tm][tn]);
    }
  }
}

template <int OP>
__global__ __launch_bounds__(256) void gemm_kernel(const hbf* __restrict__ A,
                                                   const hbf* __restrict__ Bt,
                                                   const float* __restrict__ bias,
                                                   const float* __restrict__ res,
                                                   float* __restrict__ outf,
                                                   hbf* __restrict__ outb,
                                                   int N, int K) {
  const int M = 4096;
  const int Nblk = blockIdx.x * 64;
  const int Mblk = blockIdx.y * 128;
  const int lane = threadIdx.x & 31;
  const int wave = threadIdx.x >> 5;
  const int wm = wave & 3, wn = wave >> 2;
  const int li = lane & 15, hi = lane >> 4;

  fvec8 acc[2][2] = {};
  const int nk = K / 64;

#if HAVE_TDM
  __shared__ __align__(16) char smem[2 * ATILE + 2 * BTILE];
  const u32 sOff = (u32)(uintptr_t)&smem[0];
  auto issue = [&](int stage, int t) {
    if (threadIdx.x < 32) {  // wave 0 drives the DMA (EXEC ignored by TDM)
      int kk = t * 64;
      tdm_load_2d(sOff + stage * ATILE, (const char*)A + ((size_t)Mblk * K + kk) * 2,
                  (u32)K, (u32)(M - Mblk), 64, 128, (u64)K);
      tdm_load_2d(sOff + 2 * ATILE + stage * BTILE,
                  (const char*)Bt + ((size_t)Nblk * K + kk) * 2,
                  (u32)K, (u32)(N - Nblk), 64, 64, (u64)K);
    }
  };
  issue(0, 0);
  for (int t = 0; t < nk; ++t) {
    if (t + 1 < nk) issue((t + 1) & 1, t + 1);
    if (threadIdx.x < 32) {
      if (t + 1 < nk)
        __builtin_amdgcn_s_wait_tensorcnt(1);   // next stage still in flight
      else
        __builtin_amdgcn_s_wait_tensorcnt(0);   // drain all
    }
    __syncthreads();
    gemm_tile_compute(smem + (t & 1) * ATILE, smem + 2 * ATILE + (t & 1) * BTILE,
                      wm, wn, li, hi, acc);
    __syncthreads();
  }
#else
  __shared__ __align__(16) char smem[ATILE + BTILE];
  for (int t = 0; t < nk; ++t) {
    int kk = t * 64;
    const char* gA = (const char*)A + ((size_t)Mblk * K + kk) * 2;
#pragma unroll
    for (int i = 0; i < 4; ++i) {
      int chunk = threadIdx.x + 256 * i;  // 1024 x 16B chunks, 8 per row
      int r = chunk >> 3, c = chunk & 7;
      *(uivec4*)(smem + (size_t)r * LDSP + c * 16) =
          *(const uivec4*)(gA + ((size_t)r * K + c * 8) * 2);
    }
    const char* gB = (const char*)Bt + ((size_t)Nblk * K + kk) * 2;
#pragma unroll
    for (int i = 0; i < 2; ++i) {
      int chunk = threadIdx.x + 256 * i;  // 512 chunks
      int r = chunk >> 3, c = chunk & 7;
      *(uivec4*)(smem + ATILE + (size_t)r * LDSP + c * 16) =
          *(const uivec4*)(gB + ((size_t)r * K + c * 8) * 2);
    }
    __syncthreads();
    gemm_tile_compute(smem, smem + ATILE, wm, wn, li, hi, acc);
    __syncthreads();
  }
#endif

  const int gm0 = Mblk + wm * 32;
  const int gn0 = Nblk + wn * 32;
#pragma unroll
  for (int tm = 0; tm < 2; ++tm)
#pragma unroll
    for (int tn = 0; tn < 2; ++tn) {
      int gn = gn0 + tn * 16 + li;
      float bv = bias[gn];
#pragma unroll
      for (int v = 0; v < 8; ++v) {
        int gm = gm0 + tm * 16 + v + 8 * hi;
        float val = acc[tm][tn][v] + bv;
        size_t idx = (size_t)gm * N + gn;
        if (OP == 2)
          outf[idx] = val + res[idx];
        else if (OP == 1)
          outb[idx] = __float2bfloat16(0.5f * val * (1.0f + erff(val * 0.70710678118f)));
        else
          outb[idx] = __float2bfloat16(val);
      }
    }
}

// ---------------------------------------------------------------------------
// Attention: 1 wave per 16-row q-tile of one (b,h); flash-style over 32-key
// tiles. Relative-position bias done with WMMA: U = Q x T[m0..m0+47]^T, then
// bias(l,r) = U[l][l-r+31] gathered via LDS.
// ---------------------------------------------------------------------------
__global__ __launch_bounds__(32) void attn_kernel(const hbf* __restrict__ qb,
                                                  const hbf* __restrict__ kb,
                                                  const hbf* __restrict__ vt,
                                                  const hbf* __restrict__ dist,
                                                  const float* __restrict__ mask,
                                                  hbf* __restrict__ ctx) {
  const int S = 1024, DM = 768;
  const int qt = blockIdx.x, h = blockIdx.y, b = blockIdx.z;
  const int q0 = qt * 16;
  const int lane = threadIdx.x;
  const int li = lane & 15, hi = lane >> 4;

  __shared__ float Ulds[16][48];
  __shared__ __align__(16) hbf Plds[16][32];

  Frag qa[2];
#pragma unroll
  for (int kc = 0; kc < 2; ++kc) {
    const char* p = (const char*)qb +
        ((size_t)(b * S + q0 + li) * DM + h * 64 + kc * 32 + hi * 16) * 2;
    qa[kc].u[0] = *(const uivec4*)p;
    qa[kc].u[1] = *(const uivec4*)(p + 16);
  }

  float m_s[8], l_s[8];
  fvec8 O[4] = {};
#pragma unroll
  for (int v = 0; v < 8; ++v) { m_s[v] = -3.0e38f; l_s[v] = 0.0f; }

  for (int j = 0; j < S / 32; ++j) {
    const int k0 = j * 32;
    if (j + 1 < S / 32)
      __builtin_prefetch((const char*)kb + ((size_t)(b * S + k0 + 32 + lane) * DM + h * 64) * 2, 0, 1);

    // --- raw scores Q K^T (16x32, two 16-key D tiles)
    fvec8 Sc[2];
#pragma unroll
    for (int cb = 0; cb < 2; ++cb) {
      fvec8 a = {};
#pragma unroll
      for (int kc = 0; kc < 2; ++kc) {
        Frag bfr;
        const char* p = (const char*)kb +
            ((size_t)(b * S + k0 + cb * 16 + li) * DM + h * 64 + kc * 32 + hi * 16) * 2;
        bfr.u[0] = *(const uivec4*)p;
        bfr.u[1] = *(const uivec4*)(p + 16);
        a = WMMA_BF16(qa[kc].v, bfr.v, a);
      }
      Sc[cb] = a;
    }

    // --- U = Q x T^T over the 47-wide distance window (3 x 16 cols)
    const int m0 = q0 - k0 + 992;  // = q0-k0+1023-31
#pragma unroll
    for (int cb = 0; cb < 3; ++cb) {
      int tr = m0 + cb * 16 + li;
      tr = tr < 0 ? 0 : (tr > 2046 ? 2046 : tr);
      fvec8 a = {};
#pragma unroll
      for (int kc = 0; kc < 2; ++kc) {
        Frag bfr;
        const char* p = (const char*)dist + ((size_t)tr * 64 + kc * 32 + hi * 16) * 2;
        bfr.u[0] = *(const uivec4*)p;
        bfr.u[1] = *(const uivec4*)(p + 16);
        a = WMMA_BF16(qa[kc].v, bfr.v, a);
      }
#pragma unroll
      for (int v = 0; v < 8; ++v) Ulds[v + 8 * hi][cb * 16 + li] = a[v];
    }

    // --- combine: (qk + bias) * 1/sqrt(64) + mask
    float sc[2][8];
#pragma unroll
    for (int cb = 0; cb < 2; ++cb) {
      float mk = mask[(size_t)b * S + k0 + cb * 16 + li];
#pragma unroll
      for (int v = 0; v < 8; ++v) {
        int lrow = v + 8 * hi;
        int ri = cb * 16 + li;
        float bias = Ulds[lrow][lrow - ri + 31];
        sc[cb][v] = (Sc[cb][v] + bias) * 0.125f + mk;
      }
    }

    // --- online softmax (row stats per D-layout VGPR, 16-lane reductions)
#pragma unroll
    for (int v = 0; v < 8; ++v) {
      float t = fmaxf(sc[0][v], sc[1][v]);
#pragma unroll
      for (int o = 8; o >= 1; o >>= 1) t = fmaxf(t, __shfl_xor(t, o, 16));
      float mn = fmaxf(m_s[v], t);
      float scale = __expf(m_s[v] - mn);
      float p0 = __expf(sc[0][v] - mn);
      float p1 = __expf(sc[1][v] - mn);
      sc[0][v] = p0; sc[1][v] = p1;
      float rs = p0 + p1;
#pragma unroll
      for (int o = 8; o >= 1; o >>= 1) rs += __shfl_xor(rs, o, 16);
      m_s[v] = mn;
      l_s[v] = l_s[v] * scale + rs;
#pragma unroll
      for (int t4 = 0; t4 < 4; ++t4) O[t4][v] *= scale;
    }

    // --- probs: D layout -> LDS -> A fragment (16x32 bf16)
#pragma unroll
    for (int cb = 0; cb < 2; ++cb)
#pragma unroll
      for (int v = 0; v < 8; ++v)
        Plds[v + 8 * hi][cb * 16 + li] = __float2bfloat16(sc[cb][v]);
    Frag pa;
    {
      const char* p = (const char*)&Plds[0][0] + (size_t)li * 64 + hi * 32;
      pa.u[0] = *(const uivec4*)p;
      pa.u[1] = *(const uivec4*)(p + 16);
    }

    // --- O += P x V (V pre-transposed: vt[(b*H+h)*64+d][s])
#pragma unroll
    for (int nb = 0; nb < 4; ++nb) {
      Frag bfr;
      const char* p = (const char*)vt +
          ((size_t)((b * 12 + h) * 64 + nb * 16 + li) * S + k0 + hi * 16) * 2;
      bfr.u[0] = *(const uivec4*)p;
      bfr.u[1] = *(const uivec4*)(p + 16);
      O[nb] = WMMA_BF16(pa.v, bfr.v, O[nb]);
    }
  }

#pragma unroll
  for (int nb = 0; nb < 4; ++nb)
#pragma unroll
    for (int v = 0; v < 8; ++v) {
      int row = v + 8 * hi;
      float val = O[nb][v] / l_s[v];
      ctx[(size_t)(b * S + q0 + row) * DM + h * 64 + nb * 16 + li] = __float2bfloat16(val);
    }
}

// ---------------------------------------------------------------------------
// Helper kernels
// ---------------------------------------------------------------------------
__global__ void cast_bf16_kernel(const float* __restrict__ in, hbf* __restrict__ out, int n) {
  int i = blockIdx.x * 256 + threadIdx.x;
  if (i < n) out[i] = __float2bfloat16(in[i]);
}

__global__ void cast_dual_kernel(const float* __restrict__ in, float* __restrict__ outf,
                                 hbf* __restrict__ outb, int n) {
  int i = blockIdx.x * 256 + threadIdx.x;
  if (i < n) { float v = in[i]; outf[i] = v; outb[i] = __float2bfloat16(v); }
}

// f32 [R,C] -> bf16 [C,R]
__global__ void transpose_cast_kernel(const float* __restrict__ in, hbf* __restrict__ out,
                                      int R, int C) {
  __shared__ float t[32][33];
  int c0 = blockIdx.x * 32, r0 = blockIdx.y * 32;
  int tx = threadIdx.x, ty = threadIdx.y;
  for (int i = ty; i < 32; i += 8) {
    int r = r0 + i, c = c0 + tx;
    t[i][tx] = (r < R && c < C) ? in[(size_t)r * C + c] : 0.0f;
  }
  __syncthreads();
  for (int i = ty; i < 32; i += 8) {
    int c = c0 + i, r = r0 + tx;
    if (c < C && r < R) out[(size_t)c * R + r] = __float2bfloat16(t[tx][i]);
  }
}

// vb [B*S, 768] -> vt [B, H*64, S]
__global__ void transpose_v_kernel(const hbf* __restrict__ vb, hbf* __restrict__ vt) {
  const int S = 1024, DM = 768;
  __shared__ hbf t[32][33];
  int b = blockIdx.z;
  int d0 = blockIdx.x * 32, s0 = blockIdx.y * 32;
  int tx = threadIdx.x, ty = threadIdx.y;
  for (int i = ty; i < 32; i += 8)
    t[i][tx] = vb[(size_t)(b * S + s0 + i) * DM + d0 + tx];
  __syncthreads();
  for (int i = ty; i < 32; i += 8)
    vt[(size_t)b * DM * S + (size_t)(d0 + i) * S + s0 + tx] = t[tx][i];
}

// LayerNorm over rows of 768; writes f32 + bf16 shadow
__global__ __launch_bounds__(256) void layernorm_kernel(const float* __restrict__ in,
                                                        const float* __restrict__ g,
                                                        const float* __restrict__ bb,
                                                        float* __restrict__ outf,
                                                        hbf* __restrict__ outb) {
  const int DM = 768;
  int row = blockIdx.x, t = threadIdx.x;
  const float* x = in + (size_t)row * DM;
  float v[3], s = 0.0f, s2 = 0.0f;
#pragma unroll
  for (int i = 0; i < 3; ++i) {
    v[i] = x[t + i * 256];
    s += v[i]; s2 += v[i] * v[i];
  }
  __shared__ float r1[256], r2[256];
  r1[t] = s; r2[t] = s2;
  __syncthreads();
  for (int o = 128; o > 0; o >>= 1) {
    if (t < o) { r1[t] += r1[t + o]; r2[t] += r2[t + o]; }
    __syncthreads();
  }
  float mean = r1[0] * (1.0f / DM);
  float var = r2[0] * (1.0f / DM) - mean * mean;
  float inv = rsqrtf(var + 1e-12f);
#pragma unroll
  for (int i = 0; i < 3; ++i) {
    int c = t + i * 256;
    float y = (v[i] - mean) * inv * g[c] + bb[c];
    outf[(size_t)row * DM + c] = y;
    outb[(size_t)row * DM + c] = __float2bfloat16(y);
  }
}

// ---------------------------------------------------------------------------
// Host orchestration
// ---------------------------------------------------------------------------
extern "C" void kernel_launch(void* const* d_in, const int* in_sizes, int n_in,
                              void* d_out, int out_size, void* d_ws, size_t ws_size,
                              hipStream_t stream) {
  (void)in_sizes; (void)n_in; (void)out_size; (void)ws_size;
  const int Bb = 4, S = 1024, DM = 768, H = 12, DF = 3072, L = 2;
  const size_t M = (size_t)Bb * S;  // 4096

  const float* hidden = (const float*)d_in[0];
  const float* mask   = (const float*)d_in[1];
  const float* Wq  = (const float*)d_in[2];  const float* bq  = (const float*)d_in[3];
  const float* Wk  = (const float*)d_in[4];  const float* bk  = (const float*)d_in[5];
  const float* Wv  = (const float*)d_in[6];  const float* bv  = (const float*)d_in[7];
  const float* Wao = (const float*)d_in[8];  const float* bao = (const float*)d_in[9];
  const float* ln1g = (const float*)d_in[10]; const float* ln1b = (const float*)d_in[11];
  const float* Wi  = (const float*)d_in[12]; const float* bi  = (const float*)d_in[13];
  const float* Wo2 = (const float*)d_in[14]; const float* bo2 = (const float*)d_in[15];
  const float* ln2g = (const float*)d_in[16]; const float* ln2b = (const float*)d_in[17];
  const float* dist = (const float*)d_in[18];
  float* out = (float*)d_out;

  char* ws = (char*)d_ws;
  size_t off = 0;
  auto alloc = [&](size_t bytes) -> void* {
    void* p = ws + off;
    off = (off + bytes + 255) & ~(size_t)255;
    return p;
  };
  float* xf    = (float*)alloc(M * DM * 4);
  float* attnf = (float*)alloc(M * DM * 4);
  float* pre   = (float*)alloc(M * DM * 4);
  hbf* xb    = (hbf*)alloc(M * DM * 2);
  hbf* qbuf  = (hbf*)alloc(M * DM * 2);
  hbf* kbuf  = (hbf*)alloc(M * DM * 2);
  hbf* vbuf  = (hbf*)alloc(M * DM * 2);
  hbf* ctxb  = (hbf*)alloc(M * DM * 2);
  hbf* attnb = (hbf*)alloc(M * DM * 2);
  hbf* vt    = (hbf*)alloc(M * DM * 2);
  hbf* ffb   = (hbf*)alloc(M * DF * 2);
  hbf *wqT[L], *wkT[L], *wvT[L], *waoT[L], *wiT[L], *wo2T[L];
  for (int l = 0; l < L; ++l) {
    wqT[l]  = (hbf*)alloc((size_t)DM * DM * 2);
    wkT[l]  = (hbf*)alloc((size_t)DM * DM * 2);
    wvT[l]  = (hbf*)alloc((size_t)DM * DM * 2);
    waoT[l] = (hbf*)alloc((size_t)DM * DM * 2);
    wiT[l]  = (hbf*)alloc((size_t)DM * DF * 2);
    wo2T[l] = (hbf*)alloc((size_t)DM * DF * 2);
  }
  hbf* distb = (hbf*)alloc((size_t)L * 2047 * 64 * 2);

  // --- weight prep (transpose to [N][K] bf16) + input cast
  dim3 tb(32, 8);
  for (int l = 0; l < L; ++l) {
    transpose_cast_kernel<<<dim3(24, 24), tb, 0, stream>>>(Wq  + (size_t)l * DM * DM, wqT[l],  DM, DM);
    transpose_cast_kernel<<<dim3(24, 24), tb, 0, stream>>>(Wk  + (size_t)l * DM * DM, wkT[l],  DM, DM);
    transpose_cast_kernel<<<dim3(24, 24), tb, 0, stream>>>(Wv  + (size_t)l * DM * DM, wvT[l],  DM, DM);
    transpose_cast_kernel<<<dim3(24, 24), tb, 0, stream>>>(Wao + (size_t)l * DM * DM, waoT[l], DM, DM);
    transpose_cast_kernel<<<dim3(96, 24), tb, 0, stream>>>(Wi  + (size_t)l * DM * DF, wiT[l],  DM, DF);
    transpose_cast_kernel<<<dim3(24, 96), tb, 0, stream>>>(Wo2 + (size_t)l * DF * DM, wo2T[l], DF, DM);
  }
  {
    int n = L * 2047 * 64;
    cast_bf16_kernel<<<(n + 255) / 256, 256, 0, stream>>>(dist, distb, n);
    int m = (int)(M * DM);
    cast_dual_kernel<<<(m + 255) / 256, 256, 0, stream>>>(hidden, xf, xb, m);
  }

  for (int l = 0; l < L; ++l) {
    // QKV projections
    gemm_kernel<0><<<dim3(DM / 64, M / 128), 256, 0, stream>>>(xb, wqT[l], bq + l * DM, nullptr, nullptr, qbuf, DM, DM);
    gemm_kernel<0><<<dim3(DM / 64, M / 128), 256, 0, stream>>>(xb, wkT[l], bk + l * DM, nullptr, nullptr, kbuf, DM, DM);
    gemm_kernel<0><<<dim3(DM / 64, M / 128), 256, 0, stream>>>(xb, wvT[l], bv + l * DM, nullptr, nullptr, vbuf, DM, DM);
    transpose_v_kernel<<<dim3(24, 32, 4), tb, 0, stream>>>(vbuf, vt);
    // attention with relative-position bias
    attn_kernel<<<dim3(S / 16, H, Bb), dim3(32), 0, stream>>>(qbuf, kbuf, vt,
        distb + (size_t)l * 2047 * 64, mask, ctxb);
    // out projection + residual, LN1
    gemm_kernel<2><<<dim3(DM / 64, M / 128), 256, 0, stream>>>(ctxb, waoT[l], bao + l * DM, xf, pre, nullptr, DM, DM);
    layernorm_kernel<<<(int)M, 256, 0, stream>>>(pre, ln1g + l * DM, ln1b + l * DM, attnf, attnb);
    // FFN
    gemm_kernel<1><<<dim3(DF / 64, M / 128), 256, 0, stream>>>(attnb, wiT[l], bi + l * DF, nullptr, nullptr, ffb, DF, DM);
    gemm_kernel<2><<<dim3(DM / 64, M / 128), 256, 0, stream>>>(ffb, wo2T[l], bo2 + l * DM, attnf, pre, nullptr, DM, DF);
    layernorm_kernel<<<(int)M, 256, 0, stream>>>(pre, ln2g + l * DM, ln2b + l * DM,
                                                 (l == L - 1) ? out : xf, xb);
  }
}
